// EncoderTransformer_77859167142214
// MI455X (gfx1250) — matrix operand order repaired and checked by reading, not compiled, not run on hardware
//
#include <hip/hip_runtime.h>
#include <cmath>

// ---------------------------------------------------------------------------
// CDNA5 (gfx1250) window-attention encoder.
// All GEMMs run on v_wmma_f32_16x16x32_f16 with f16 operands staged into LDS
// via global_load_async_to_lds_b128/b64 (ASYNCcnt), double-buffered so copies
// for step i+1 overlap the WMMAs of step i. Operands are produced in f16,
// halving GEMM read traffic (the net is HBM-bandwidth bound on MI455X).
// ---------------------------------------------------------------------------

typedef _Float16 v16h __attribute__((ext_vector_type(16)));
typedef float    v8f  __attribute__((ext_vector_type(8)));

union FragAB {
    v16h     v;
    float4   f4[2];
    _Float16 h[16];
};

__device__ __forceinline__ float gelu_exact(float x) {
    return 0.5f * x * (1.0f + erff(x * 0.70710678118654752f));
}

// Async copies global -> LDS (per-lane addresses), tracked by ASYNCcnt.
// Generic LDS pointers carry the LDS byte offset in their low 32 bits.
__device__ __forceinline__ void async_copy_b128(void* lds_dst, const void* gsrc) {
    unsigned lds = (unsigned)(size_t)lds_dst;
    asm volatile("global_load_async_to_lds_b128 %0, %1, off"
                 :: "v"(lds), "v"(gsrc) : "memory");
}
__device__ __forceinline__ void async_copy_b64(void* lds_dst, const void* gsrc) {
    unsigned lds = (unsigned)(size_t)lds_dst;
    asm volatile("global_load_async_to_lds_b64 %0, %1, off"
                 :: "v"(lds), "v"(gsrc) : "memory");
}
// wait until ASYNCcnt <= 3: newest batch (3 ops/thread) may still be in
// flight, the previous batch is guaranteed complete.
__device__ __forceinline__ void wait_async_le3() {
    asm volatile("s_wait_asynccnt 0x3" ::: "memory");
}
__device__ __forceinline__ void wait_async_le0() {
    asm volatile("s_wait_asynccnt 0x0" ::: "memory");
}

// ---------------------------------------------------------------------------
// Token-major WMMA GEMM:  C[M,N] = A[M,K] (f16,row-major) @ Bw[N,K]^T (f16)
// Requires: M % 128 == 0, N % 32 == 0, K % 32 == 0 (true for every call here).
// Block: 256 threads = 8 waves as 4(M) x 2(N); wave tile 32x16 (2 WMMA/step).
// LDS double-buffered; each thread issues exactly 3 async ops per K-step
// (2x b128 for A, 1x b64 for B) so ASYNCcnt thresholds are uniform.
// ---------------------------------------------------------------------------
__global__ __launch_bounds__(256)
void gemm_tok_wmma(const _Float16* __restrict__ A,
                   const _Float16* __restrict__ Bw,
                   void* __restrict__ Cout, int outHalf,
                   int M, int N, int K, int doGelu)
{
    __shared__ __align__(16) _Float16 As[2][128][40];  // pad keeps 16B align
    __shared__ __align__(16) _Float16 Bs[2][32][40];

    const int tid  = threadIdx.x;
    const int lane = tid & 31;
    const int hlf  = lane >> 4;
    const int lm   = lane & 15;
    const int wid  = tid >> 5;
    const int wm   = (wid >> 1) * 32;   // 0,32,64,96
    const int wn   = (wid & 1) * 16;    // 0,16

    const int m0 = blockIdx.y * 128;
    const int n0 = blockIdx.x * 32;

    // A tile 128x32 halfs: 2x b128 per thread
    const int ar = tid >> 1;            // 0..127
    const int ac = (tid & 1) * 16;      // 0 or 16
    const _Float16* agp = A + (long)(m0 + ar) * K + ac;
    // B tile 32x32 halfs: 1x b64 per thread (all 256 threads)
    const int br = tid >> 3;            // 0..31
    const int bc = (tid & 7) * 4;       // 0,4,..,28
    const _Float16* bgp = Bw + (long)(n0 + br) * K + bc;

    // prologue: stage step 0 into buffer 0
    async_copy_b128(&As[0][ar][ac],     agp);
    async_copy_b128(&As[0][ar][ac + 8], agp + 8);
    async_copy_b64 (&Bs[0][br][bc],     bgp);

    v8f acc0 = {0.f,0.f,0.f,0.f,0.f,0.f,0.f,0.f};
    v8f acc1 = {0.f,0.f,0.f,0.f,0.f,0.f,0.f,0.f};

    const int steps = K >> 5;
    for (int i = 0; i < steps; ++i) {
        const int cb = i & 1;
        if (i + 1 < steps) {            // prefetch step i+1 into other buffer
            const int kk = (i + 1) << 5;
            const int nb = cb ^ 1;
            async_copy_b128(&As[nb][ar][ac],     agp + kk);
            async_copy_b128(&As[nb][ar][ac + 8], agp + kk + 8);
            async_copy_b64 (&Bs[nb][br][bc],     bgp + kk);
            wait_async_le3();           // step i's batch complete
        } else {
            wait_async_le0();
        }
        __syncthreads();

        // A frags: lane half 0 -> K 0..7 & 16..23 ; half 1 -> K 8..15 & 24..31
        FragAB a0, a1, b;
        const _Float16* p0 = &As[cb][wm + lm][hlf * 8];
        a0.f4[0] = *(const float4*)(p0);
        a0.f4[1] = *(const float4*)(p0 + 16);
        const _Float16* p1 = &As[cb][wm + 16 + lm][hlf * 8];
        a1.f4[0] = *(const float4*)(p1);
        a1.f4[1] = *(const float4*)(p1 + 16);
        // B frag: lane half 0 -> K 0..7 & 16..23 ; half 1 -> K 8..15 & 24..31
        const _Float16* pb = &Bs[cb][wn + lm][hlf * 8];
        b.f4[0] = *(const float4*)(pb);
        b.f4[1] = *(const float4*)(pb + 16);

        acc0 = __builtin_amdgcn_wmma_f32_16x16x32_f16(
            false, a0.v, false, b.v, (short)0, acc0, false, false);
        acc1 = __builtin_amdgcn_wmma_f32_16x16x32_f16(
            false, a1.v, false, b.v, (short)0, acc1, false, false);
        __syncthreads();                // protect buffer cb before step i+2 issue
    }

    const int n = n0 + wn + lm;
#pragma unroll
    for (int r = 0; r < 8; ++r) {
        const int mA = m0 + wm + r + hlf * 8;
        const int mB = mA + 16;
        float v0 = acc0[r], v1 = acc1[r];
        if (doGelu) { v0 = gelu_exact(v0); v1 = gelu_exact(v1); }
        if (outHalf) {
            _Float16* C16 = (_Float16*)Cout;
            C16[(long)mA * N + n] = (_Float16)v0;
            C16[(long)mB * N + n] = (_Float16)v1;
        } else {
            float* C32 = (float*)Cout;
            C32[(long)mA * N + n] = v0;
            C32[(long)mB * N + n] = v1;
        }
    }
}

// ---------------------------------------------------------------------------
// One wave per (window, head). 16 tokens/window, dh in {4,8}.
// QK^T and PV each one zero-padded 16x16x32 WMMA; shuffle softmax.
// Consumes f16 qkv, emits f16 (feeds the out-proj GEMM directly).
// ---------------------------------------------------------------------------
__global__ __launch_bounds__(32)
void window_attn_kernel(const _Float16* __restrict__ qkv,
                        const float* __restrict__ rel_emb,
                        _Float16* __restrict__ outb,
                        int C, int heads, int dh, float scale)
{
    const int win  = blockIdx.x / heads;
    const int h    = blockIdx.x % heads;
    const int lane = threadIdx.x;
    const int hlf  = lane >> 4;
    const int lm   = lane & 15;
    const long base = (long)win * 16;
    const int C3 = 3 * C;

    __shared__ __align__(16) _Float16 attn_s[16][24];

    FragAB qf, kf;
#pragma unroll
    for (int e = 0; e < 16; ++e) {
        const int kq = (hlf ? 8 : 0) + (e < 8 ? e : e + 8);  // A layout K index
        float qv = 0.f;
        if (kq < dh) qv = (float)qkv[(base + lm) * C3 + h * dh + kq] * scale;
        qf.h[e] = (_Float16)qv;
        const int kk = (hlf ? 8 : 0) + (e < 8 ? e : e + 8);  // B layout K index
        kf.h[e] = (kk < dh) ? qkv[(base + lm) * C3 + C + h * dh + kk]
                            : (_Float16)0.f;
    }
    v8f sim = {0.f,0.f,0.f,0.f,0.f,0.f,0.f,0.f};
    sim = __builtin_amdgcn_wmma_f32_16x16x32_f16(
        false, qf.v, false, kf.v, (short)0, sim, false, false);

    float rowv[8];
#pragma unroll
    for (int r = 0; r < 8; ++r) {
        const int i = r + hlf * 8, j = lm;
        const int idx = ((i >> 2) - (j >> 2) + 3) * 7 + ((i & 3) - (j & 3) + 3);
        rowv[r] = sim[r] + rel_emb[idx * heads + h];
    }
#pragma unroll
    for (int r = 0; r < 8; ++r) {
        float mx = rowv[r];
        for (int mk = 1; mk <= 8; mk <<= 1)
            mx = fmaxf(mx, __shfl_xor(mx, mk, 32));
        const float ex = __expf(rowv[r] - mx);
        float sm = ex;
        for (int mk = 1; mk <= 8; mk <<= 1)
            sm += __shfl_xor(sm, mk, 32);
        rowv[r] = ex / sm;
    }
#pragma unroll
    for (int r = 0; r < 8; ++r)
        attn_s[r + hlf * 8][lm] = (_Float16)rowv[r];
    __syncthreads();

    FragAB af, vf;
#pragma unroll
    for (int e = 0; e < 16; ++e) {
        const int ka = (hlf ? 8 : 0) + (e < 8 ? e : e + 8);
        af.h[e] = (ka < 16) ? attn_s[lm][ka] : (_Float16)0.f;
        const int kj = (hlf ? 8 : 0) + (e < 8 ? e : e + 8);  // key token for V, pad K 16->32
        vf.h[e] = (kj < 16 && lm < dh)
                    ? qkv[(base + kj) * C3 + 2 * C + h * dh + lm]
                    : (_Float16)0.f;
    }
    v8f o = {0.f,0.f,0.f,0.f,0.f,0.f,0.f,0.f};
    o = __builtin_amdgcn_wmma_f32_16x16x32_f16(
        false, af.v, false, vf.v, (short)0, o, false, false);

    if (lm < dh) {
#pragma unroll
        for (int r = 0; r < 8; ++r) {
            const int i = r + hlf * 8;
            outb[(base + i) * C + h * dh + lm] = (_Float16)o[r];
        }
    }
}

// ---------------------------------------------------------------------------
// Elementwise / conv kernels
// ---------------------------------------------------------------------------
__global__ void cvt_f32_f16_kernel(const float* __restrict__ s,
                                   _Float16* __restrict__ d, long n)
{
    const long t = (long)blockIdx.x * blockDim.x + threadIdx.x;
    if (t < n) d[t] = (_Float16)s[t];
}

__global__ void conv2d_kernel(const float* __restrict__ x, const float* __restrict__ w,
                              const float* __restrict__ bias, float* __restrict__ y,
                              int B, int Cin, int H, int W, int Cout,
                              int Kk, int stride, int pad, int Ho, int Wo)
{
    const long t = (long)blockIdx.x * blockDim.x + threadIdx.x;
    const long total = (long)B * Cout * Ho * Wo;
    if (t >= total) return;
    const int wo = (int)(t % Wo);
    long r = t / Wo;
    const int ho = (int)(r % Ho); r /= Ho;
    const int co = (int)(r % Cout);
    const int bb = (int)(r / Cout);
    float acc = bias[co];
    for (int ci = 0; ci < Cin; ++ci)
        for (int kh = 0; kh < Kk; ++kh) {
            const int ih = ho * stride - pad + kh;
            if (ih < 0 || ih >= H) continue;
            for (int kw = 0; kw < Kk; ++kw) {
                const int iw = wo * stride - pad + kw;
                if (iw < 0 || iw >= W) continue;
                acc += x[(((long)bb * Cin + ci) * H + ih) * W + iw] *
                       w[(((long)co * Cin + ci) * Kk + kh) * Kk + kw];
            }
        }
    y[(((long)bb * Cout + co) * Ho + ho) * Wo + wo] = acc;
}

__global__ void ln_cf_kernel(const float* __restrict__ x, const float* __restrict__ w,
                             const float* __restrict__ b, float* __restrict__ y,
                             int C, long HW, long total /* B*HW */)
{
    const long t = (long)blockIdx.x * blockDim.x + threadIdx.x;
    if (t >= total) return;
    const long bi = t / HW, hw = t - bi * HW;
    const float* xp = x + bi * (long)C * HW + hw;
    float s = 0.f, s2 = 0.f;
    for (int c = 0; c < C; ++c) { const float v = xp[(long)c * HW]; s += v; s2 += v * v; }
    const float mean = s / (float)C;
    const float var  = s2 / (float)C - mean * mean;
    const float inv  = rsqrtf(var + 1e-6f);
    float* yp = y + bi * (long)C * HW + hw;
    for (int c = 0; c < C; ++c)
        yp[(long)c * HW] = w[c] * (xp[(long)c * HW] - mean) * inv + b[c];
}

// channels-first LN -> token-major f16 (feeds the MLP GEMMs)
__global__ void ln_cf_to_tok_h_kernel(const float* __restrict__ x,
                                      const float* __restrict__ w,
                                      const float* __restrict__ b,
                                      _Float16* __restrict__ tok,
                                      int C, long HW, long total /* B*HW */)
{
    const long t = (long)blockIdx.x * blockDim.x + threadIdx.x;
    if (t >= total) return;
    const long bi = t / HW, hw = t - bi * HW;
    const float* xp = x + bi * (long)C * HW + hw;
    float s = 0.f, s2 = 0.f;
    for (int c = 0; c < C; ++c) { const float v = xp[(long)c * HW]; s += v; s2 += v * v; }
    const float mean = s / (float)C;
    const float var  = s2 / (float)C - mean * mean;
    const float inv  = rsqrtf(var + 1e-6f);
    _Float16* tp = tok + t * C;
    for (int c = 0; c < C; ++c)
        tp[c] = (_Float16)(w[c] * (xp[(long)c * HW] - mean) * inv + b[c]);
}

__global__ void dwconv_gelu_res_kernel(const float* __restrict__ xln,
                                       const float* __restrict__ skip,
                                       const float* __restrict__ w,
                                       float* __restrict__ y,
                                       int B, int C, int H, int W)
{
    const long t = (long)blockIdx.x * blockDim.x + threadIdx.x;
    const long total = (long)B * C * H * W;
    if (t >= total) return;
    const int ww = (int)(t % W);
    long r = t / W;
    const int hh = (int)(r % H); r /= H;
    const int c  = (int)(r % C);
    const long bc = t / ((long)H * W);
    float acc = 0.f;
    for (int kh = 0; kh < 3; ++kh) {
        int ih = hh - 1 + kh; ih = ih < 0 ? 0 : (ih >= H ? H - 1 : ih);  // replicate
        for (int kw = 0; kw < 3; ++kw) {
            int iw = ww - 1 + kw; iw = iw < 0 ? 0 : (iw >= W ? W - 1 : iw);
            acc += xln[(bc * H + ih) * W + iw] * w[c * 9 + kh * 3 + kw];
        }
    }
    y[t] = skip[t] + gelu_exact(acc);
}

__global__ void deform_offsets_kernel(const float* __restrict__ x /* batch0 (C,H,W) */,
                                      const float* __restrict__ wo,
                                      const float* __restrict__ bo,
                                      int* __restrict__ xg, int* __restrict__ yg,
                                      int C, int H, int W)
{
    const int t = blockIdx.x * blockDim.x + threadIdx.x;
    if (t >= H * W) return;
    const int ww = t % W, hh = t / W;
    const float max_off = (float)(H > W ? H : W) * 0.25f;
    float off[2];
    for (int oc = 0; oc < 2; ++oc) {
        float acc = bo[oc];
        for (int c = 0; c < C; ++c)
            for (int kh = 0; kh < 3; ++kh) {
                const int ih = hh - 1 + kh;
                if (ih < 0 || ih >= H) continue;
                for (int kw = 0; kw < 3; ++kw) {
                    const int iw = ww - 1 + kw;
                    if (iw < 0 || iw >= W) continue;
                    acc += x[((long)c * H + ih) * W + iw] *
                           wo[(((long)oc * C + c) * 3 + kh) * 3 + kw];
                }
            }
        off[oc] = fminf(fmaxf(acc, -max_off), max_off);
    }
    int gx = (int)((float)hh + off[0]);  // trunc toward zero == astype(int32)
    int gy = (int)((float)ww + off[1]);
    if (gx < 0) gx += H; else if (gx >= H) gx = H - 1;
    if (gy < 0) gy += W; else if (gy >= W) gy = W - 1;
    xg[t] = gx; yg[t] = gy;
}

// fused deform gather + strided-grid window partition, emits f16 token-major
// h = w1*Hx + xh, w = w2*Wy + yw  (w1/w2 outer);  token = w1*4 + w2
__global__ void deform_window_pack_kernel(const float* __restrict__ x,
                                          const int* __restrict__ xg,
                                          const int* __restrict__ yg,
                                          _Float16* __restrict__ xw,
                                          int B, int C, int H, int W)
{
    const long t = (long)blockIdx.x * blockDim.x + threadIdx.x;
    const long total = (long)B * C * H * W;
    if (t >= total) return;
    const int ww = (int)(t % W);
    long r = t / W;
    const int hh = (int)(r % H); r /= H;
    const int c  = (int)(r % C);
    const int b  = (int)(r / C);
    const long bc = t / ((long)H * W);
    const int hw = hh * W + ww;
    const float val = x[(bc * H + xg[hw]) * W + yg[hw]];
    const int Hx = H / 4, Wy = W / 4;
    const int w1 = hh / Hx, xh = hh % Hx;
    const int w2 = ww / Wy, yw = ww % Wy;
    const long m = (((long)b * Hx + xh) * Wy + yw) * 16 + w1 * 4 + w2;
    xw[m * C + c] = (_Float16)val;
}

__global__ void window_unpack_res_kernel(const float* __restrict__ proj,
                                         const float* __restrict__ ls1,
                                         const float* __restrict__ skip,
                                         float* __restrict__ y,
                                         int B, int C, int H, int W)
{
    const long t = (long)blockIdx.x * blockDim.x + threadIdx.x;
    const long total = (long)B * C * H * W;
    if (t >= total) return;
    const int ww = (int)(t % W);
    long r = t / W;
    const int hh = (int)(r % H); r /= H;
    const int c  = (int)(r % C);
    const int b  = (int)(r / C);
    const int Hx = H / 4, Wy = W / 4;
    const int w1 = hh / Hx, xh = hh % Hx;
    const int w2 = ww / Wy, yw = ww % Wy;
    const long m = (((long)b * Hx + xh) * Wy + yw) * 16 + w1 * 4 + w2;
    y[t] = ls1[c] * proj[m * C + c] + skip[t];
}

// x_cf + ls[c] * mlp_tok  ->  y_cf   (token m = b*HW + hw, plain order)
__global__ void tok_scale_add_cf_kernel(const float* __restrict__ xcf,
                                        const float* __restrict__ mtok,
                                        const float* __restrict__ ls,
                                        float* __restrict__ y,
                                        int C, long HW, long total)
{
    const long t = (long)blockIdx.x * blockDim.x + threadIdx.x;
    if (t >= total) return;
    const long hw = t % HW;
    const long c  = (t / HW) % C;
    const long bi = t / (HW * C);
    y[t] = xcf[t] + ls[c] * mtok[(bi * HW + hw) * C + c];
}

// ---------------------------------------------------------------------------
// Host orchestration. Leaf order (jax pytree, dict keys sorted):
//   0: x
//   1..8  : norm[i].{b,w}
//   9..24 : pe[i].{b,lnb,lnw,w}
//   25..  : stages[s][blk].{ln0b,ln0w,ln1b,ln1w,ln2b,ln2w,ls1,ls2,
//                            mlp_w1,mlp_w2,off_b,off_w,out_w,pos_w,qkv_w,rel_emb}
// ---------------------------------------------------------------------------
extern "C" void kernel_launch(void* const* d_in, const int* in_sizes, int n_in,
                              void* d_out, int out_size, void* d_ws, size_t ws_size,
                              hipStream_t stream)
{
    (void)in_sizes; (void)n_in; (void)out_size; (void)ws_size;
    static const int DIMS_[4] = {32, 64, 128, 256};
    static const int DHD[4]   = {4, 4, 8, 8};
    static const int PEK[4]   = {7, 3, 3, 3};
    static const int PES[4]   = {4, 2, 2, 2};
    const int Bn = 16;

    auto in = [&](int i) { return (const float*)d_in[i]; };

    const long ACT = (long)Bn * 32 * 64 * 64;  // max activation elems (stage 0)
    float* ws   = (float*)d_ws;
    float* actA = ws;                 // f32 activation ping
    float* actB = actA + ACT;         // f32 activation pong
    float* t1   = actB + ACT;         // f32 scratch (LN out, conv out)
    float* t2   = t1 + ACT;           // f32 scratch (mlp token out)
    float* pjB  = t2 + ACT;           // f32 (B*HW, C) out-proj
    _Float16* qkv_h = (_Float16*)(pjB + ACT);   // f16 (B*HW, 3C)
    _Float16* xw_h  = qkv_h + 3 * ACT;          // f16 (B*HW, C) packed windows
    _Float16* ao_h  = xw_h + ACT;               // f16 attention out
    _Float16* tok_h = ao_h + ACT;               // f16 LN2 token-major
    _Float16* hid_h = tok_h + ACT;              // f16 MLP hidden (B*HW, 4C)
    _Float16* wb_h  = hid_h + 4 * ACT;          // f16 weight scratch (<= 4C*C)
    int* xg = (int*)(wb_h + 4 * 256 * 256);
    int* yg = xg + 64 * 64;

    int H = 256, W = 256, Cin = 3;
    const float* cur = (const float*)d_in[0];
    long outOff = 0;

    auto cvt = [&](const float* s, _Float16* d, long n) {
        cvt_f32_f16_kernel<<<(unsigned)((n + 255) / 256), 256, 0, stream>>>(s, d, n);
    };

    for (int s = 0; s < 4; ++s) {
        const int C = DIMS_[s];
        const int K = PEK[s], st = PES[s], pd = K / 2;
        const int Ho = (H + 2 * pd - K) / st + 1;
        const int Wo = (W + 2 * pd - K) / st + 1;
        const float *pe_b   = in(9 + 4 * s),  *pe_lnb = in(10 + 4 * s);
        const float *pe_lnw = in(11 + 4 * s), *pe_w   = in(12 + 4 * s);
        {
            const long tot = (long)Bn * C * Ho * Wo;
            conv2d_kernel<<<(unsigned)((tot + 255) / 256), 256, 0, stream>>>(
                cur, pe_w, pe_b, t1, Bn, Cin, H, W, C, K, st, pd, Ho, Wo);
        }
        H = Ho; W = Wo;
        const long HW = (long)H * W;
        const long total = (long)Bn * C * HW, BHW = (long)Bn * HW;
        ln_cf_kernel<<<(unsigned)((BHW + 255) / 256), 256, 0, stream>>>(
            t1, pe_lnw, pe_lnb, actB, C, HW, BHW);

        float* bufs[2] = {actA, actB};
        int cx = 1;
        const int heads = C / DHD[s], dh = DHD[s];
        const float scale = 1.0f / sqrtf((float)dh);
        const int C3 = 3 * C, C4 = 4 * C;

        for (int blk = 0; blk < 2; ++blk) {
            const int pb = 25 + (s * 2 + blk) * 16;
            const float *ln0b = in(pb + 0),  *ln0w = in(pb + 1);
            const float *ln1b = in(pb + 2),  *ln1w = in(pb + 3);
            const float *ln2b = in(pb + 4),  *ln2w = in(pb + 5);
            const float *ls1  = in(pb + 6),  *ls2  = in(pb + 7);
            const float *mw1  = in(pb + 8),  *mw2  = in(pb + 9);
            const float *offb = in(pb + 10), *offw = in(pb + 11);
            const float *outw = in(pb + 12), *posw = in(pb + 13);
            const float *qkvw = in(pb + 14), *relE = in(pb + 15);

            float* x = bufs[cx];
            // --- pos conv: x = x + gelu(dwconv3(ln0(x))) ---------------------
            ln_cf_kernel<<<(unsigned)((BHW + 255) / 256), 256, 0, stream>>>(
                x, ln0w, ln0b, t1, C, HW, BHW);
            dwconv_gelu_res_kernel<<<(unsigned)((total + 255) / 256), 256, 0, stream>>>(
                t1, x, posw, bufs[1 - cx], Bn, C, H, W);
            cx ^= 1; x = bufs[cx];

            // --- window attention -------------------------------------------
            ln_cf_kernel<<<(unsigned)((BHW + 255) / 256), 256, 0, stream>>>(
                x, ln1w, ln1b, t1, C, HW, BHW);
            deform_offsets_kernel<<<(unsigned)((H * W + 255) / 256), 256, 0, stream>>>(
                t1, offw, offb, xg, yg, C, H, W);
            deform_window_pack_kernel<<<(unsigned)((total + 255) / 256), 256, 0, stream>>>(
                t1, xg, yg, xw_h, Bn, C, H, W);
            cvt(qkvw, wb_h, (long)C3 * C);
            {   // qkv = xw @ qkv_w^T  (f16 output -> attention)
                dim3 g(C3 / 32, (unsigned)(BHW / 128));
                gemm_tok_wmma<<<g, 256, 0, stream>>>(
                    xw_h, wb_h, qkv_h, 1, (int)BHW, C3, C, 0);
            }
            window_attn_kernel<<<(unsigned)((BHW / 16) * heads), 32, 0, stream>>>(
                qkv_h, relE, ao_h, C, heads, dh, scale);
            cvt(outw, wb_h, (long)C * C);
            {   // proj = attn_out @ out_w^T
                dim3 g(C / 32, (unsigned)(BHW / 128));
                gemm_tok_wmma<<<g, 256, 0, stream>>>(
                    ao_h, wb_h, pjB, 0, (int)BHW, C, C, 0);
            }
            window_unpack_res_kernel<<<(unsigned)((total + 255) / 256), 256, 0, stream>>>(
                pjB, ls1, x, bufs[1 - cx], Bn, C, H, W);
            cx ^= 1; x = bufs[cx];   // x = attention residual "out"

            // --- MLP (token-major) ------------------------------------------
            ln_cf_to_tok_h_kernel<<<(unsigned)((BHW + 255) / 256), 256, 0, stream>>>(
                x, ln2w, ln2b, tok_h, C, HW, BHW);
            cvt(mw1, wb_h, (long)C4 * C);
            {   // hid = gelu(tok @ mlp_w1^T), f16 output
                dim3 g(C4 / 32, (unsigned)(BHW / 128));
                gemm_tok_wmma<<<g, 256, 0, stream>>>(
                    tok_h, wb_h, hid_h, 1, (int)BHW, C4, C, 1);
            }
            cvt(mw2, wb_h, (long)C * C4);
            {   // mlp_out = hid @ mlp_w2^T, f32 token-major
                dim3 g(C / 32, (unsigned)(BHW / 128));
                gemm_tok_wmma<<<g, 256, 0, stream>>>(
                    hid_h, wb_h, t2, 0, (int)BHW, C, C4, 0);
            }
            tok_scale_add_cf_kernel<<<(unsigned)((total + 255) / 256), 256, 0, stream>>>(
                x, t2, ls2, bufs[1 - cx], C, HW, total);
            cx ^= 1;
        }

        // --- stage norm into d_out; also feeds next patch embed --------------
        const float *nb = in(1 + 2 * s), *nw = in(2 + 2 * s);
        float* outp = (float*)d_out + outOff;
        ln_cf_kernel<<<(unsigned)((BHW + 255) / 256), 256, 0, stream>>>(
            bufs[cx], nw, nb, outp, C, HW, BHW);
        outOff += total;
        cur = outp;
        Cin = C;
    }
}
